// SR_GNN_9019431322288
// MI455X (gfx1250) — compile-verified, matrix operand order checked by reading, not run on hardware
//
#include <hip/hip_runtime.h>
#include <hip/hip_bf16.h>

typedef __bf16 bf16;
typedef __attribute__((ext_vector_type(16))) __bf16 v16bf;
typedef __attribute__((ext_vector_type(8)))  __bf16 v8bf;
typedef __attribute__((ext_vector_type(4)))  __bf16 v4bf;
typedef __attribute__((ext_vector_type(8)))  float  v8f;
typedef __attribute__((ext_vector_type(4)))  unsigned int v4u;
typedef __attribute__((ext_vector_type(4)))  int v4i;
typedef __attribute__((ext_vector_type(8)))  int v8i;

#define HID    256
#define NNODES 8192
#define NSESS  512
#define NITEMS 100000

#if defined(__gfx1250__) && __has_builtin(__builtin_amdgcn_tensor_load_to_lds)
#define HAVE_TDM 1
#else
#define HAVE_TDM 0
#endif

#if HAVE_TDM
// ---------------------------------------------------------------------------
// Issue one TDM DMA of a 2D bf16 tile [rows x 64] (row stride = ld elements)
// from global into LDS at byte offset lds_off. D# layout per ISA ch.8:
//  group0: count=1 | lds_addr | global_addr[56:0] | type=2
//  group1: data_size=1(2B), tensor_dim0=64, tensor_dim1=rows,
//          tile_dim0=64, tile_dim1=rows, tensor_dim0_stride=ld
// Tracked with TENSORcnt; caller waits s_wait_tensorcnt 0.
// ---------------------------------------------------------------------------
static __device__ __forceinline__ void tdm_load_tile_bf16(
    unsigned long long gaddr, unsigned int lds_off, int rows, int ld) {
  v4u g0;
  g0[0] = 1u;                                   // count=1, user mode
  g0[1] = lds_off;                              // lds_addr (bytes)
  g0[2] = (unsigned int)gaddr;                  // global_addr[31:0]
  g0[3] = (unsigned int)(gaddr >> 32) | (2u << 30);  // addr[56:32] | type=2
  v8i g1;
  g1[0] = 0x00010000;                           // data_size=1 -> 2 bytes
  g1[1] = (int)(64u << 16);                     // tensor_dim0[15:0]=64 @bit48
  g1[2] = (int)(((unsigned)rows) << 16);        // tensor_dim1[15:0]=rows @bit80
  g1[3] = (int)(64u << 16);                     // tile_dim0=64 @bit112
  g1[4] = rows;                                 // tile_dim1=rows @bit128
  g1[5] = ld;                                   // tensor_dim0_stride[31:0] @bit160
  g1[6] = 0;
  g1[7] = 0;
  v4i zz = {0, 0, 0, 0};
#if __clang_major__ >= 23
  v8i zz8 = {0, 0, 0, 0, 0, 0, 0, 0};
  __builtin_amdgcn_tensor_load_to_lds(g0, g1, zz, zz, zz8, 0);
#else
  __builtin_amdgcn_tensor_load_to_lds(g0, g1, zz, zz, 0);
#endif
}
#endif

// ---------------------------------------------------------------------------
// WMMA fragment load from LDS. 16-bit A/B operand layout (ISA 7.12.2):
//   lanes 0-15  : K = 0..7  (VGPR0-3) and K = 16..23 (VGPR4-7)
//   lanes 16-31 : K = 8..15           and K = 24..31
// ---------------------------------------------------------------------------
static __device__ __forceinline__ v16bf frag_from_lds(const bf16* base, int half) {
  v8bf lo = *(const v8bf*)(base + half * 8);
  v8bf hi = *(const v8bf*)(base + 16 + half * 8);
  v16bf f;
#pragma unroll
  for (int i = 0; i < 8; ++i) { f[i] = lo[i]; f[i + 8] = hi[i]; }
  return f;
}

// ---------------------------------------------------------------------------
// Generic bf16 WMMA GEMM:  C[M x Ncols] = A[M x K] * B[K x Ncols]
// B supplied transposed ("BT", [Ncols x K] row-major). Block tile 128x64,
// BK=64, 256 threads = 8 waves (4x2), wave tile 32x32 = 2x2 WMMA fragments.
// AF32: A is fp32, converted to bf16 while staging (VALU path).
// bf16 A tiles and all B tiles are staged by the Tensor Data Mover.
// EPI:  0 = +bias, bf16 transposed   1 = relu, bf16 row-major
//       2 = fp32 + bf16 row-major    3 = +bias, fp32 row-major
//       4 = +bias, bf16 row-major    5 = fp32 transposed, guard row < M
// ---------------------------------------------------------------------------
template<bool AF32, int EPI>
__global__ __launch_bounds__(256) void gemm_wmma(
    const void* __restrict__ Aptr, int lda,
    const bf16* __restrict__ BT, int ldb,
    int M, int K,
    const float* __restrict__ bias,
    float* __restrict__ outF, bf16* __restrict__ outB, bf16* __restrict__ outBT,
    int ldo)
{
  __shared__ bf16 ldsA[128 * 64];   // 16 KB
  __shared__ bf16 ldsB[64 * 64];    //  8 KB

  const int tid  = threadIdx.x;
  const int lane = tid & 31;
  const int wave = tid >> 5;
  const int wm   = wave >> 1;
  const int wn   = wave & 1;
  const int half = lane >> 4;
  const int l16  = lane & 15;

  const int bm = blockIdx.y * 128;
  const int bn = blockIdx.x * 64;

  const float* Af = (const float*)Aptr;
  const bf16*  Ab = (const bf16*)Aptr;

  v8f acc[2][2] = {};

  for (int k0 = 0; k0 < K; k0 += 64) {
    __syncthreads();

#if HAVE_TDM
    // ---- TDM staging: wave 0 issues DMA(s), waits TENSORcnt ----
    if (wave == 0) {
      unsigned long long bga =
          (unsigned long long)(size_t)(BT + (size_t)bn * ldb + k0);
      tdm_load_tile_bf16(bga, (unsigned int)(size_t)&ldsB[0], 64, ldb);
      if constexpr (!AF32) {
        unsigned long long aga =
            (unsigned long long)(size_t)(Ab + (size_t)bm * lda + k0);
        tdm_load_tile_bf16(aga, (unsigned int)(size_t)&ldsA[0], 128, lda);
      }
      __builtin_amdgcn_s_wait_tensorcnt(0);
    }
    if constexpr (AF32) {
#pragma unroll
      for (int it = 0; it < 8; ++it) {
        int li  = tid + it * 256;
        int row = li >> 4;
        int c4  = (li & 15) * 4;
        int ar  = bm + row; if (ar > M - 1) ar = M - 1;
        const float* src = Af + (size_t)ar * lda + k0 + c4;
        if (k0 + 64 < K) __builtin_prefetch(src + 64, 0, 1);
        float4 v = *(const float4*)src;
        v4bf p; p[0] = (bf16)v.x; p[1] = (bf16)v.y; p[2] = (bf16)v.z; p[3] = (bf16)v.w;
        *(v4bf*)(ldsA + row * 64 + c4) = p;
      }
    }
#else
    // ---- fallback staging (also what the host pass parses) ----
    if constexpr (AF32) {
#pragma unroll
      for (int it = 0; it < 8; ++it) {
        int li  = tid + it * 256;
        int row = li >> 4;
        int c4  = (li & 15) * 4;
        int ar  = bm + row; if (ar > M - 1) ar = M - 1;
        const float* src = Af + (size_t)ar * lda + k0 + c4;
        float4 v = *(const float4*)src;
        v4bf p; p[0] = (bf16)v.x; p[1] = (bf16)v.y; p[2] = (bf16)v.z; p[3] = (bf16)v.w;
        *(v4bf*)(ldsA + row * 64 + c4) = p;
      }
    } else {
#pragma unroll
      for (int it = 0; it < 4; ++it) {
        int li  = tid + it * 256;
        int row = li >> 3;
        int c8  = (li & 7) * 8;
        *(v8bf*)(ldsA + row * 64 + c8) =
            *(const v8bf*)(Ab + (size_t)(bm + row) * lda + k0 + c8);
      }
    }
#pragma unroll
    for (int it = 0; it < 2; ++it) {
      int li = tid + it * 256;
      int n  = li >> 3;
      int c8 = (li & 7) * 8;
      *(v8bf*)(ldsB + n * 64 + c8) =
          *(const v8bf*)(BT + (size_t)(bn + n) * ldb + k0 + c8);
    }
#endif
    __syncthreads();

#pragma unroll
    for (int kk = 0; kk < 2; ++kk) {
      v16bf afrag[2], bfrag[2];
#pragma unroll
      for (int tm = 0; tm < 2; ++tm)
        afrag[tm] = frag_from_lds(ldsA + (wm * 32 + tm * 16 + l16) * 64 + kk * 32, half);
#pragma unroll
      for (int tn = 0; tn < 2; ++tn)
        bfrag[tn] = frag_from_lds(ldsB + (wn * 32 + tn * 16 + l16) * 64 + kk * 32, half);
#pragma unroll
      for (int tm = 0; tm < 2; ++tm)
#pragma unroll
        for (int tn = 0; tn < 2; ++tn)
          acc[tm][tn] = __builtin_amdgcn_wmma_f32_16x16x32_bf16(
              false, afrag[tm], false, bfrag[tn], (short)0, acc[tm][tn], false, false);
    }
  }

  // ---- epilogue. C/D layout: VGPR r, lanes 0-15 -> M=r, lanes 16-31 -> M=r+8
#pragma unroll
  for (int tm = 0; tm < 2; ++tm) {
#pragma unroll
    for (int tn = 0; tn < 2; ++tn) {
      const int gcol = bn + wn * 32 + tn * 16 + l16;
#pragma unroll
      for (int r = 0; r < 8; ++r) {
        const int grow = bm + wm * 32 + tm * 16 + half * 8 + r;
        float v = acc[tm][tn][r];
        if constexpr (EPI == 0) {
          v += bias[gcol];
          outBT[(size_t)gcol * M + grow] = (bf16)v;
        } else if constexpr (EPI == 1) {
          v = v > 0.f ? v : 0.f;
          outB[(size_t)grow * ldo + gcol] = (bf16)v;
        } else if constexpr (EPI == 2) {
          outF[(size_t)grow * ldo + gcol] = v;
          outB[(size_t)grow * ldo + gcol] = (bf16)v;
        } else if constexpr (EPI == 3) {
          v += bias[gcol];
          outF[(size_t)grow * ldo + gcol] = v;
        } else if constexpr (EPI == 4) {
          v += bias[gcol];
          outB[(size_t)grow * ldo + gcol] = (bf16)v;
        } else {                      // EPI == 5
          if (grow < M) outF[(size_t)gcol * ldo + grow] = v;
        }
      }
    }
  }
}

// ---------------------------------------------------------------------------
// Small elementwise / gather / reduce kernels
// ---------------------------------------------------------------------------
__global__ void cvt_bf16(const float* __restrict__ in, bf16* __restrict__ out, int n) {
  int i = blockIdx.x * 256 + threadIdx.x;
  if (i < n) out[i] = (bf16)in[i];
}

__global__ void gather_rows_bf16(const float* __restrict__ src, const int* __restrict__ rows,
                                 bf16* __restrict__ dst, int n) {
  int i = blockIdx.x * 256 + threadIdx.x;
  if (i >= n) return;
  int r = i >> 8, h = i & 255;
  dst[i] = (bf16)src[(size_t)rows[r] * HID + h];
}

__global__ __launch_bounds__(256) void alpha_kernel(
    const float* __restrict__ q1, const float* __restrict__ q2,
    const int* __restrict__ idx, const float* __restrict__ Wf3,
    float* __restrict__ alpha) {
  __shared__ float red[256];
  const int n = blockIdx.x, t = threadIdx.x;
  const int s = idx[n];
  float v = q1[(size_t)s * HID + t] + q2[(size_t)n * HID + t];
  v = 1.f / (1.f + __expf(-v));
  red[t] = v * Wf3[t];
  __syncthreads();
  for (int off = 128; off > 0; off >>= 1) {
    if (t < off) red[t] += red[t + off];
    __syncthreads();
  }
  if (t == 0) alpha[n] = red[0];
}

__global__ void zero_f32(float* p, int n) {
  int i = blockIdx.x * 256 + threadIdx.x;
  if (i < n) p[i] = 0.f;
}

__global__ void scatter_add(const float* __restrict__ z, const float* __restrict__ alpha,
                            const int* __restrict__ idx, float* __restrict__ sg) {
  int i = blockIdx.x * 256 + threadIdx.x;
  int n = i >> 8, h = i & 255;
  atomicAdd(&sg[(size_t)idx[n] * HID + h], alpha[n] * z[i]);
}

__global__ void build_s(const bf16* __restrict__ slb, const float* __restrict__ sg,
                        bf16* __restrict__ sb) {
  int i = blockIdx.x * 256 + threadIdx.x;
  int b = i >> 9, c = i & 511;
  sb[i] = (c < 256) ? slb[(size_t)b * HID + c] : (bf16)sg[(size_t)b * HID + (c - 256)];
}

__global__ __launch_bounds__(256) void softmax_rows(float* __restrict__ out, int n) {
  __shared__ float sm[256], ss[256];
  float* row = out + (size_t)blockIdx.x * n;
  const int t = threadIdx.x;
  float m = -3.4e38f, s = 0.f;
  for (int i = t; i < n; i += 256) {
    float x = row[i];
    float nm = fmaxf(m, x);
    s = s * __expf(m - nm) + __expf(x - nm);
    m = nm;
  }
  sm[t] = m; ss[t] = s;
  __syncthreads();
  for (int off = 128; off > 0; off >>= 1) {
    if (t < off) {
      float m2 = sm[t + off], s2 = ss[t + off];
      float nm = fmaxf(sm[t], m2);
      ss[t] = ss[t] * __expf(sm[t] - nm) + s2 * __expf(m2 - nm);
      sm[t] = nm;
    }
    __syncthreads();
  }
  const float Mx = sm[0], invS = 1.f / ss[0];
  for (int i = t; i < n; i += 256) row[i] = __expf(row[i] - Mx) * invS;
}

// ---------------------------------------------------------------------------
extern "C" void kernel_launch(void* const* d_in, const int* in_sizes, int n_in,
                              void* d_out, int out_size, void* d_ws, size_t ws_size,
                              hipStream_t stream) {
  const float* adj  = (const float*)d_in[0];
  const int*   items= (const int*)  d_in[1];
  const int*   last = (const int*)  d_in[2];
  const int*   idx  = (const int*)  d_in[3];
  const float* emb  = (const float*)d_in[4];
  const float* W1   = (const float*)d_in[5];  const float* b1  = (const float*)d_in[6];
  const float* W2   = (const float*)d_in[7];  const float* b2  = (const float*)d_in[8];
  const float* Wf1  = (const float*)d_in[9];  const float* bf1 = (const float*)d_in[10];
  const float* Wf2  = (const float*)d_in[11]; const float* bf2 = (const float*)d_in[12];
  const float* Wf3  = (const float*)d_in[13];
  const float* Wf4  = (const float*)d_in[14]; const float* bf4 = (const float*)d_in[15];
  float* out = (float*)d_out;
  (void)in_sizes; (void)n_in; (void)out_size; (void)ws_size;

  char* p = (char*)d_ws;
  auto alloc = [&](size_t bytes) { char* r = p; p += (bytes + 255) & ~(size_t)255; return r; };
  bf16*  Xb    = (bf16*)alloc((size_t)NNODES * HID * 2);
  bf16*  W1b   = (bf16*)alloc((size_t)HID * HID * 2);
  bf16*  W2b   = (bf16*)alloc((size_t)HID * HID * 2);
  bf16*  Wf1b  = (bf16*)alloc((size_t)HID * HID * 2);
  bf16*  Wf2b  = (bf16*)alloc((size_t)HID * HID * 2);
  bf16*  Wf4b  = (bf16*)alloc((size_t)HID * 2 * HID * 2);
  bf16*  t1T   = (bf16*)alloc((size_t)HID * NNODES * 2);
  bf16*  hB    = (bf16*)alloc((size_t)NNODES * HID * 2);
  bf16*  t2T   = (bf16*)alloc((size_t)HID * NNODES * 2);
  float* z     = (float*)alloc((size_t)NNODES * HID * 4);
  bf16*  zb    = (bf16*)alloc((size_t)NNODES * HID * 2);
  bf16*  slb   = (bf16*)alloc((size_t)NSESS * HID * 2);
  float* q1pre = (float*)alloc((size_t)NSESS * HID * 4);
  float* q2    = (float*)alloc((size_t)NNODES * HID * 4);
  float* alpha = (float*)alloc((size_t)NNODES * 4);
  float* sg    = (float*)alloc((size_t)NSESS * HID * 4);
  bf16*  sb    = (bf16*)alloc((size_t)NSESS * 2 * HID * 2);
  bf16*  shb   = (bf16*)alloc((size_t)NSESS * HID * 2);

  const int HH = HID * HID;
  cvt_bf16<<<(HH + 255) / 256, 256, 0, stream>>>(W1,  W1b,  HH);
  cvt_bf16<<<(HH + 255) / 256, 256, 0, stream>>>(W2,  W2b,  HH);
  cvt_bf16<<<(HH + 255) / 256, 256, 0, stream>>>(Wf1, Wf1b, HH);
  cvt_bf16<<<(HH + 255) / 256, 256, 0, stream>>>(Wf2, Wf2b, HH);
  cvt_bf16<<<(2 * HH + 255) / 256, 256, 0, stream>>>(Wf4, Wf4b, 2 * HH);

  gather_rows_bf16<<<NNODES, 256, 0, stream>>>(emb, items, Xb, NNODES * HID);

  gemm_wmma<false, 0><<<dim3(HID / 64, NNODES / 128), 256, 0, stream>>>(
      Xb, HID, W1b, HID, NNODES, HID, b1, nullptr, nullptr, t1T, 0);
  gemm_wmma<true, 1><<<dim3(HID / 64, NNODES / 128), 256, 0, stream>>>(
      adj, NNODES, t1T, NNODES, NNODES, NNODES, nullptr, nullptr, hB, nullptr, HID);
  gemm_wmma<false, 0><<<dim3(HID / 64, NNODES / 128), 256, 0, stream>>>(
      hB, HID, W2b, HID, NNODES, HID, b2, nullptr, nullptr, t2T, 0);
  gemm_wmma<true, 2><<<dim3(HID / 64, NNODES / 128), 256, 0, stream>>>(
      adj, NNODES, t2T, NNODES, NNODES, NNODES, nullptr, z, zb, nullptr, HID);

  gather_rows_bf16<<<NSESS, 256, 0, stream>>>(z, last, slb, NSESS * HID);
  gemm_wmma<false, 3><<<dim3(HID / 64, NSESS / 128), 256, 0, stream>>>(
      slb, HID, Wf1b, HID, NSESS, HID, bf1, q1pre, nullptr, nullptr, HID);
  gemm_wmma<false, 3><<<dim3(HID / 64, NNODES / 128), 256, 0, stream>>>(
      zb, HID, Wf2b, HID, NNODES, HID, bf2, q2, nullptr, nullptr, HID);

  alpha_kernel<<<NNODES, 256, 0, stream>>>(q1pre, q2, idx, Wf3, alpha);

  zero_f32<<<(NSESS * HID + 255) / 256, 256, 0, stream>>>(sg, NSESS * HID);
  scatter_add<<<NNODES, 256, 0, stream>>>(z, alpha, idx, sg);

  build_s<<<(NSESS * 2 * HID) / 256, 256, 0, stream>>>(slb, sg, sb);
  gemm_wmma<false, 4><<<dim3(HID / 64, NSESS / 128), 256, 0, stream>>>(
      sb, 2 * HID, Wf4b, 2 * HID, NSESS, 2 * HID, bf4, nullptr, shb, nullptr, HID);

  gemm_wmma<true, 5><<<dim3(NSESS / 64, (NITEMS + 127) / 128), 256, 0, stream>>>(
      emb, HID, shb, HID, NITEMS, HID, nullptr, out, nullptr, nullptr, NITEMS);

  softmax_rows<<<NSESS, 256, 0, stream>>>(out, NITEMS);
}
